// NODE_20306605375982
// MI455X (gfx1250) — matrix operand order repaired and checked by reading, not compiled, not run on hardware
//
#include <hip/hip_runtime.h>
#include <math.h>

// Problem constants (match reference)
#define N_PTS 16384
#define DDIM  64
#define HDIM  128
#define T_PTS 20
#define MAXSTEPS 4
#define RTOL_C 1e-6f
#define ATOL_C 1e-7f

typedef __attribute__((ext_vector_type(16))) _Float16 v16h;
typedef __attribute__((ext_vector_type(8)))  _Float16 v8h;
typedef __attribute__((ext_vector_type(8)))  float    v8f;
typedef __attribute__((ext_vector_type(4)))  unsigned int v4u;
typedef __attribute__((ext_vector_type(8)))  unsigned int v8u;

// ---------------- LDS layout (bytes) for the step kernel ----------------
// [0, 66816) is staged by ONE tensor_load_to_lds from the pre-packed block in ws.
#define W1T_OFF 0          // 128x64  f16 : W1t[h][d]      (16384 B)
#define W2T_OFF 16384      // 128x128 f16 : W2t[n][k]      (32768 B)
#define W3T_OFF 49152      // 64x128  f16 : W3t[n][k]      (16384 B)
#define B1_OFF  65536      // 128 f32
#define B2_OFF  66048      // 128 f32
#define B3_OFF  66560      // 64  f32
#define WGT_BYTES 66816    // packed weight block size (= 16704 dwords)
#define WGT_DW   16704
#define XB_OFF  66816      // 4 waves x 16x64  f16         (8192 B)
#define HB_OFF  75008      // 4 waves x 16x128 f16         (16384 B)
#define KB_OFF  91392      // 4 waves x 6x16x64 f32        (98304 B)
#define RED_OFF 189696     // 128 f32
#define SMEM_BYTES 190208

// Dormand-Prince tableau (row 6 == b-row, so the stage-7 input IS y5)
__constant__ float c_AA[7][6] = {
  {0.f, 0.f, 0.f, 0.f, 0.f, 0.f},
  {1.f/5.f, 0.f, 0.f, 0.f, 0.f, 0.f},
  {3.f/40.f, 9.f/40.f, 0.f, 0.f, 0.f, 0.f},
  {44.f/45.f, -56.f/15.f, 32.f/9.f, 0.f, 0.f, 0.f},
  {19372.f/6561.f, -25360.f/2187.f, 64448.f/6561.f, -212.f/729.f, 0.f, 0.f},
  {9017.f/3168.f, -355.f/33.f, 46732.f/5247.f, 49.f/176.f, -5103.f/18656.f, 0.f},
  {35.f/384.f, 0.f, 500.f/1113.f, 125.f/192.f, -2187.f/6784.f, 11.f/84.f}};
__constant__ float c_EE[7] = {
  (float)(35.0/384.0 - 5179.0/57600.0), 0.f,
  (float)(500.0/1113.0 - 7571.0/16695.0),
  (float)(125.0/192.0 - 393.0/640.0),
  (float)(-2187.0/6784.0 + 92097.0/339200.0),
  (float)(11.0/84.0 - 187.0/2100.0),
  (float)(-1.0/40.0)};

__device__ __forceinline__ float fast_tanh(float x) {
#if __has_builtin(__builtin_amdgcn_tanhf)
  return __builtin_amdgcn_tanhf(x);   // v_tanh_f32 on gfx1250
#else
  float e = __expf(2.0f * x);
  return 1.0f - 2.0f / (e + 1.0f);
#endif
}

// A-fragment (16x32 f16): lane (lo,hi) holds row M=lo, K = [kt*32+16*hi .. +15]
__device__ __forceinline__ v16h load_afrag(const _Float16* A, int ldA, int lo, int hi, int kt) {
  const _Float16* p = A + lo * ldA + kt * 32 + 16 * hi;
  v8h c0 = *(const v8h*)p;
  v8h c1 = *(const v8h*)(p + 8);
  v16h r;
#pragma unroll
  for (int i = 0; i < 8; ++i) { r[i] = c0[i]; r[i + 8] = c1[i]; }
  return r;
}

// Bt stored [N][K] f16 (row = one output column, contiguous K).
// B-fragment: lane (lo,hi) holds N = nt*16+lo, K = kt*32+16*hi+i -> one aligned 32B LDS read.
// First K-tile uses C = 0 (SRC2 inline 0); bias is folded into the epilogue.
template <int KT, int NT>
__device__ __forceinline__ void gemm(const _Float16* A, int ldA,
                                     const _Float16* Bt, int ldB,
                                     int lo, int hi, v8f* acc) {
  const v8f z = {0.f, 0.f, 0.f, 0.f, 0.f, 0.f, 0.f, 0.f};
  {
    v16h af = load_afrag(A, ldA, lo, hi, 0);
#pragma unroll
    for (int nt = 0; nt < NT; ++nt) {
      v16h bf = *(const v16h*)(Bt + (nt * 16 + lo) * ldB + 16 * hi);
      acc[nt] = __builtin_amdgcn_wmma_f32_16x16x32_f16(
          false, af, false, bf, (short)0, z, false, false);
    }
  }
#pragma unroll
  for (int kt = 1; kt < KT; ++kt) {
    v16h af = load_afrag(A, ldA, lo, hi, kt);
#pragma unroll
    for (int nt = 0; nt < NT; ++nt) {
      v16h bf = *(const v16h*)(Bt + (nt * 16 + lo) * ldB + kt * 32 + 16 * hi);
      acc[nt] = __builtin_amdgcn_wmma_f32_16x16x32_f16(
          false, af, false, bf, (short)0, acc[nt], false, false);
    }
  }
}

template <int NT>
__device__ __forceinline__ void act_store(const v8f* acc, const float* bias,
                                          _Float16* H, int ldH, int lo, int hi) {
#pragma unroll
  for (int nt = 0; nt < NT; ++nt) {
    float bv = bias[nt * 16 + lo];
#pragma unroll
    for (int r = 0; r < 8; ++r)
      H[(r + 8 * hi) * ldH + nt * 16 + lo] = (_Float16)fast_tanh(acc[nt][r] + bv);
  }
}

// ---------------- one dopri5 trial step over all rows ----------------
__global__ __launch_bounds__(128)
void node_step(const float* __restrict__ sc,
               const float* __restrict__ Yg, float* __restrict__ Y5g,
               float* __restrict__ partials,
               const unsigned int* __restrict__ wgt) {
  extern __shared__ char smem[];
  _Float16* w1t = (_Float16*)(smem + W1T_OFF);
  _Float16* w2t = (_Float16*)(smem + W2T_OFF);
  _Float16* w3t = (_Float16*)(smem + W3T_OFF);
  float* b1s = (float*)(smem + B1_OFF);
  float* b2s = (float*)(smem + B2_OFF);
  float* b3s = (float*)(smem + B3_OFF);
  float* red = (float*)(smem + RED_OFF);

  const int tid  = threadIdx.x;
  const int wave = tid >> 5, lane = tid & 31, lo = lane & 15, hi = lane >> 4;
  _Float16* xb = (_Float16*)(smem + XB_OFF) + wave * 16 * 64;    // stage input (A of layer1)
  _Float16* hb = (_Float16*)(smem + HB_OFF) + wave * 16 * 128;   // hidden acts (A of layer2/3)
  float*    kb = (float*)(smem + KB_OFF)   + wave * 6 * 16 * 64; // k1..k6 (f32)

  // ---- stage packed weights+biases via the Tensor Data Mover (one DMA) ----
  if (wave == 0) {
    unsigned long long ga = (unsigned long long)(uintptr_t)wgt;
    unsigned int lds0 = (unsigned int)(size_t)(void*)smem; // LDS byte address (low 32 bits)
    v4u g0;
    g0[0] = 1u;                                   // count=1, user descriptor
    g0[1] = lds0;                                 // lds_addr
    g0[2] = (unsigned int)ga;                     // global_addr[31:0]
    g0[3] = (unsigned int)((ga >> 32) & 0x01FFFFFFu) | (2u << 30); // addr[56:32] | type=2
    v8u g1;
    g1[0] = 2u << 16;                             // wg_mask=0, data_size=4B, no flags
    g1[1] = ((unsigned)WGT_DW & 0xFFFFu) << 16;   // tensor_dim0[15:0]
    g1[2] = ((unsigned)WGT_DW >> 16) | (1u << 16);// tensor_dim0[31:16] | tensor_dim1=1
    g1[3] = ((unsigned)WGT_DW & 0xFFFFu) << 16;   // tile_dim0
    g1[4] = 1u;                                   // tile_dim1=1, tile_dim2=0
    g1[5] = (unsigned)WGT_DW;                     // tensor_dim0_stride[31:0]
    g1[6] = ((unsigned)WGT_DW & 0xFFFFu) << 16;   // stride0[47:32]=0 | stride1[15:0]
    g1[7] = ((unsigned)WGT_DW >> 16);             // stride1[47:16]
    asm volatile("tensor_load_to_lds %0, %1"
                 :
                 : "s"(g0), "s"(g1)
                 : "memory");
    __builtin_amdgcn_s_wait_tensorcnt(0);
  }
  __syncthreads();

  const float t = sc[0], dt = sc[1], t1 = sc[2];
  const float dtc = fminf(dt, t1 - t);     // never overshoot

  const int rowBase = blockIdx.x * 64 + wave * 16;
  v8f y[4], ea[4], y5f[4];
#pragma unroll
  for (int nt = 0; nt < 4; ++nt) {
#pragma unroll
    for (int r = 0; r < 8; ++r) {
      y[nt][r] = Yg[(size_t)(rowBase + r + 8 * hi) * DDIM + nt * 16 + lo];
      ea[nt][r] = 0.0f;
    }
    y5f[nt] = y[nt];
  }

  for (int s = 0; s < 7; ++s) {
    // stage input  xs = y + dtc * sum_{j<s} A[s][j] * k_j   (elementwise in fragment layout)
    v8f xs[4];
#pragma unroll
    for (int nt = 0; nt < 4; ++nt) xs[nt] = y[nt];
    for (int j = 0; j < s; ++j) {
      float cj = dtc * c_AA[s][j];
#pragma unroll
      for (int nt = 0; nt < 4; ++nt)
#pragma unroll
        for (int r = 0; r < 8; ++r)
          xs[nt][r] += cj * kb[(j * 16 + r + 8 * hi) * 64 + nt * 16 + lo];
    }
#pragma unroll
    for (int nt = 0; nt < 4; ++nt)
#pragma unroll
      for (int r = 0; r < 8; ++r)
        xb[(r + 8 * hi) * 64 + nt * 16 + lo] = (_Float16)xs[nt][r];
    if (s == 6) {               // tableau row 6 is the b-row -> stage input == y5
#pragma unroll
      for (int nt = 0; nt < 4; ++nt) y5f[nt] = xs[nt];
    }

    // 3-layer MLP on the WMMA path (zero-C start, bias in epilogue)
    v8f acc8[8];
    gemm<2, 8>(xb, 64,  w1t, 64,  lo, hi, acc8);        // 16 wmma
    act_store<8>(acc8, b1s, hb, 128, lo, hi);
    gemm<4, 8>(hb, 128, w2t, 128, lo, hi, acc8);        // 32 wmma
    act_store<8>(acc8, b2s, hb, 128, lo, hi);
    v8f kc[4];
    gemm<4, 4>(hb, 128, w3t, 128, lo, hi, kc);          // 16 wmma
#pragma unroll
    for (int nt = 0; nt < 4; ++nt) {
      float b3v = b3s[nt * 16 + lo];
#pragma unroll
      for (int r = 0; r < 8; ++r) kc[nt][r] += b3v;
    }

    if (s < 6) {
#pragma unroll
      for (int nt = 0; nt < 4; ++nt)
#pragma unroll
        for (int r = 0; r < 8; ++r)
          kb[(s * 16 + r + 8 * hi) * 64 + nt * 16 + lo] = kc[nt][r];
    }
    float es = c_EE[s];
#pragma unroll
    for (int nt = 0; nt < 4; ++nt)
#pragma unroll
      for (int r = 0; r < 8; ++r)
        ea[nt][r] += es * kc[nt][r];
  }

  // error contribution + store y5
  float lsum = 0.0f;
#pragma unroll
  for (int nt = 0; nt < 4; ++nt) {
#pragma unroll
    for (int r = 0; r < 8; ++r) {
      float errv = dtc * ea[nt][r];
      float yv = y[nt][r], y5v = y5f[nt][r];
      float scv = ATOL_C + RTOL_C * fmaxf(fabsf(yv), fabsf(y5v));
      float q = errv / scv;
      lsum += q * q;
      Y5g[(size_t)(rowBase + r + 8 * hi) * DDIM + nt * 16 + lo] = y5v;
    }
  }
  red[tid] = lsum;
  __syncthreads();
#pragma unroll
  for (int off = 64; off > 0; off >>= 1) {
    if (tid < off) red[tid] += red[tid + off];
    __syncthreads();
  }
  if (tid == 0) partials[blockIdx.x] = red[0];
}

// ---------------- weight pre-pack: f32 -> f16 transposed [N][K] + f32 biases ----------------
__global__ void node_prep(const float* __restrict__ W1, const float* __restrict__ b1,
                          const float* __restrict__ W2, const float* __restrict__ b2,
                          const float* __restrict__ W3, const float* __restrict__ b3,
                          char* __restrict__ wgt) {
  int idx = blockIdx.x * blockDim.x + threadIdx.x;
  _Float16* hw = (_Float16*)wgt;
  if (idx < 8192) {                       // W1t[h][d] = W1[d][h]
    int h = idx >> 6, d = idx & 63;
    hw[idx] = (_Float16)W1[d * 128 + h];
  } else if (idx < 24576) {               // W2t[n][k] = W2[k][n]
    int j = idx - 8192; int n = j >> 7, k = j & 127;
    hw[idx] = (_Float16)W2[k * 128 + n];
  } else if (idx < 32768) {               // W3t[n][k] = W3[k][n]
    int j = idx - 24576; int n = j >> 7, k = j & 127;
    hw[idx] = (_Float16)W3[k * 64 + n];
  } else if (idx < 33088) {               // biases (f32) at byte 65536
    int j = idx - 32768;
    float v = (j < 128) ? b1[j] : (j < 256) ? b2[j - 128] : b3[j - 256];
    ((float*)(wgt + 65536))[j] = v;
  }
}

// ---------------- scalar controller: deterministic reduce + accept/dt update ----------------
__global__ void node_control(float* sc, const float* __restrict__ partials) {
  __shared__ float r2[256];
  int tid = threadIdx.x;
  r2[tid] = partials[tid];
  __syncthreads();
  for (int off = 128; off > 0; off >>= 1) {
    if (tid < off) r2[tid] += r2[tid + off];
    __syncthreads();
  }
  if (tid == 0) {
    float errn = sqrtf(r2[0] * (1.0f / (float)(N_PTS * DDIM)));
    float t = sc[0], dt = sc[1], t1 = sc[2], span = sc[3];
    bool  done   = t >= t1 - 1e-7f * span;
    float dtc    = fminf(dt, t1 - t);
    bool  accept = (errn <= 1.0f) && (!done);
    sc[4] = accept ? 1.0f : 0.0f;
    if (accept) sc[0] = t + dtc;
    float fac = 0.9f * powf(fmaxf(errn, 1e-10f), -0.2f);
    fac = fminf(fmaxf(fac, 0.2f), 10.0f);
    sc[1] = done ? dt : dtc * fac;
  }
}

__global__ void node_commit(float* __restrict__ y, const float* __restrict__ y5,
                            const float* __restrict__ sc) {
  int idx = blockIdx.x * blockDim.x + threadIdx.x;
  if (sc[4] != 0.0f) y[idx] = y5[idx];
}

__global__ void node_outwrite(float* __restrict__ out, const float* __restrict__ y, int ti) {
  int idx = blockIdx.x * blockDim.x + threadIdx.x;
  int n = idx / DDIM, d = idx % DDIM;
  out[((size_t)n * T_PTS + ti) * DDIM + d] = y[idx];
}

__global__ void node_init(float* __restrict__ out, float* __restrict__ y,
                          const float* __restrict__ x0) {
  int idx = blockIdx.x * blockDim.x + threadIdx.x;
  int n = idx / DDIM, d = idx % DDIM;
  float v = x0[idx];
  y[idx] = v;
  out[((size_t)n * T_PTS) * DDIM + d] = v;   // traj[:,0,:] = x0
}

__global__ void node_interval_init(float* sc, const float* __restrict__ tarr, int i) {
  float t0 = tarr[i], t1 = tarr[i + 1];
  sc[0] = t0;        // t
  sc[1] = t1 - t0;   // dt  (init = span)
  sc[2] = t1;        // t1
  sc[3] = t1 - t0;   // span
  sc[4] = 0.0f;      // accept flag
}

extern "C" void kernel_launch(void* const* d_in, const int* in_sizes, int n_in,
                              void* d_out, int out_size, void* d_ws, size_t ws_size,
                              hipStream_t stream) {
  (void)in_sizes; (void)n_in; (void)out_size; (void)ws_size;
  const float* t  = (const float*)d_in[0];
  const float* x0 = (const float*)d_in[1];
  const float* W1 = (const float*)d_in[2];
  const float* b1 = (const float*)d_in[3];
  const float* W2 = (const float*)d_in[4];
  const float* b2 = (const float*)d_in[5];
  const float* W3 = (const float*)d_in[6];
  const float* b3 = (const float*)d_in[7];
  float* out = (float*)d_out;

  char*  ws       = (char*)d_ws;
  float* sc       = (float*)ws;                                     // scalars
  float* partials = (float*)(ws + 256);                             // 256 block partials
  float* y        = (float*)(ws + 4096);                            // committed state [N,64]
  float* y5       = (float*)(ws + 4096 + (size_t)N_PTS * DDIM * 4); // trial state
  char*  wgt      = ws + 4096 + (size_t)N_PTS * DDIM * 8;           // packed f16 weights + f32 biases

  node_prep<<<(33088 + 255) / 256, 256, 0, stream>>>(W1, b1, W2, b2, W3, b3, wgt);

  const dim3 eg((N_PTS * DDIM) / 256);
  node_init<<<eg, 256, 0, stream>>>(out, y, x0);

  for (int i = 0; i < T_PTS - 1; ++i) {
    node_interval_init<<<1, 1, 0, stream>>>(sc, t, i);
    for (int s = 0; s < MAXSTEPS; ++s) {
      node_step<<<N_PTS / 64, 128, SMEM_BYTES, stream>>>(sc, y, y5, partials,
                                                         (const unsigned int*)wgt);
      node_control<<<1, 256, 0, stream>>>(sc, partials);
      node_commit<<<eg, 256, 0, stream>>>(y, y5, sc);
    }
    node_outwrite<<<eg, 256, 0, stream>>>(out, y, i + 1);
  }
}
